// GNN_65635690218035
// MI455X (gfx1250) — matrix operand order repaired and checked by reading, not compiled, not run on hardware
//
#include <hip/hip_runtime.h>

// GNN layer: agg = segment-mean(x[src] -> dst), h=[x|agg], out = relu(h@W+b)
// N=100000 nodes, 96 in-dim, K=192, HIDDEN=1024, 800000 edges.
//
// GEMM done as split-precision bf16 WMMA: f32 = hi(bf16) + lo(bf16),
// h@W ~= hh@Wh + hh@Wl + hl@Wh  (error ~2^-16 relative, ~fp32 quality)
// using v_wmma_f32_16x16x32_bf16 (wave32, CDNA5).

#define N_FEAT   96
#define H_FEAT   192      // 2*N_FEAT (GEMM K)
#define HIDDEN_N 1024     // GEMM N

typedef __bf16 bf16_t;
typedef __attribute__((ext_vector_type(16))) __bf16 v16bf;
typedef __attribute__((ext_vector_type(8)))  float  v8f;

// ---------------- aggregation ----------------

__global__ __launch_bounds__(256) void gnn_zero(float* __restrict__ p, int n) {
  int i = blockIdx.x * 256 + threadIdx.x;
  if (i < n) p[i] = 0.0f;
}

__global__ __launch_bounds__(256) void gnn_deg(const int* __restrict__ dst,
                                               float* __restrict__ deg, int ne) {
  int i = blockIdx.x * 256 + threadIdx.x;
  if (i < ne) atomicAdd(&deg[dst[i]], 1.0f);
}

// one thread per (edge, feature): loads and atomics are contiguous per edge row
__global__ __launch_bounds__(256) void gnn_scatter(const float* __restrict__ x,
                                                   const int* __restrict__ src,
                                                   const int* __restrict__ dst,
                                                   float* __restrict__ ns, int total) {
  int i = blockIdx.x * 256 + threadIdx.x;
  if (i >= total) return;
  int e = i / N_FEAT;
  int d = i - e * N_FEAT;
  float v = x[(size_t)src[e] * N_FEAT + d];
  atomicAdd(&ns[(size_t)dst[e] * N_FEAT + d], v);
}

// h[n] = [ x[n] | deg>0 ? ns[n]/deg : x[n] ]   (fp32, K=192 row-major)
__global__ __launch_bounds__(256) void gnn_finalize(const float* __restrict__ x,
                                                    const float* __restrict__ ns,
                                                    const float* __restrict__ deg,
                                                    float* __restrict__ h, int total) {
  int i = blockIdx.x * 256 + threadIdx.x;
  if (i >= total) return;
  int n = i / N_FEAT;
  int d = i - n * N_FEAT;
  float xv = x[i];
  float dg = deg[n];
  float a  = (dg > 0.0f) ? (ns[i] / dg) : xv;
  size_t ro = (size_t)n * H_FEAT;
  h[ro + d]          = xv;
  h[ro + N_FEAT + d] = a;
}

// W[k][n] (row-major 192x1024) -> transposed bf16 hi/lo: Wt[n][k] so B fragments
// are contiguous 32B per lane.
__global__ __launch_bounds__(256) void gnn_wprep(const float* __restrict__ W,
                                                 bf16_t* __restrict__ wh,
                                                 bf16_t* __restrict__ wl, int total) {
  int i = blockIdx.x * 256 + threadIdx.x;
  if (i >= total) return;
  int k = i >> 10;               // / HIDDEN_N
  int n = i & (HIDDEN_N - 1);
  float w  = W[i];
  bf16_t h = (bf16_t)w;          // RNE
  float  r = w - (float)h;
  size_t o = (size_t)n * H_FEAT + k;
  wh[o] = h;
  wl[o] = (bf16_t)r;
}

// ---------------- WMMA GEMM: out = relu(h @ W + b) ----------------
// Block = 256 threads = 8 waves. Each wave owns one 16-row M-strip and sweeps
// all 64 N-tiles (16 cols each), keeping all A fragments resident in VGPRs.
//
// A fragment (16x32 bf16, contiguous k order, same as B): lane half = lane>>4, m = lane&15,
//   elements 0..7  <- h[m][k0 + half*16 + 0..7]
//   elements 8..15 <- h[m][k0 + half*16 + 8..15]
// B fragment (32x16 bf16, column-major Wt): lane col n = lane&15,
//   elements 0..15 <- Wt[n][k0 + half*16 + 0..15]   (one contiguous 32B run)
// C (16x16 f32): VGPR v, m = v + 8*half, n = lane&15.

__global__ __launch_bounds__(256) void gnn_gemm(const float*  __restrict__ h,
                                                const bf16_t* __restrict__ wh,
                                                const bf16_t* __restrict__ wl,
                                                const float*  __restrict__ bias,
                                                float*        __restrict__ out,
                                                int mtiles) {
  const int wave  = threadIdx.x >> 5;
  const int lane  = threadIdx.x & 31;
  const int mtile = blockIdx.x * 8 + wave;
  if (mtile >= mtiles) return;           // wave-uniform: EXEC stays all-1s
  const int half = lane >> 4;
  const int mrow = lane & 15;

  const float* hrow = h + (size_t)(mtile * 16 + mrow) * H_FEAT;

  // Load + split all A fragments for K=192 (6 steps of K32), hi/lo bf16.
  v16bf a_hi[6], a_lo[6];
#pragma unroll
  for (int ks = 0; ks < 6; ++ks) {
    const int kf = ks * 32 + half * 16;
    const float4 f0 = *(const float4*)(hrow + kf);
    const float4 f1 = *(const float4*)(hrow + kf + 4);
    const float4 f2 = *(const float4*)(hrow + kf + 8);
    const float4 f3 = *(const float4*)(hrow + kf + 12);
    const float v[16] = {f0.x, f0.y, f0.z, f0.w, f1.x, f1.y, f1.z, f1.w,
                         f2.x, f2.y, f2.z, f2.w, f3.x, f3.y, f3.z, f3.w};
#pragma unroll
    for (int j = 0; j < 16; ++j) {
      bf16_t hi = (bf16_t)v[j];
      float  r  = v[j] - (float)hi;
      a_hi[ks][j] = hi;
      a_lo[ks][j] = (bf16_t)r;
    }
  }

  const int   bofs = mrow * H_FEAT + half * 16;     // per-lane B offset within tile
  float* orow = out + (size_t)(mtile * 16 + 8 * half) * HIDDEN_N + mrow;

  for (int nt = 0; nt < HIDDEN_N / 16; ++nt) {
    const int n0 = nt * 16;
    const bf16_t* bh_p = wh + (size_t)n0 * H_FEAT + bofs;
    const bf16_t* bl_p = wl + (size_t)n0 * H_FEAT + bofs;
    if (nt + 1 < HIDDEN_N / 16) {
      __builtin_prefetch(bh_p + (size_t)16 * H_FEAT, 0, 1);  // next B tile -> global_prefetch_b8
    }
    v8f c = {};
#pragma unroll
    for (int ks = 0; ks < 6; ++ks) {
      const v16bf bh = *(const v16bf*)(bh_p + ks * 32);
      const v16bf bl = *(const v16bf*)(bl_p + ks * 32);
      c = __builtin_amdgcn_wmma_f32_16x16x32_bf16(false, a_hi[ks], false, bh, (short)0, c, false, false);
      c = __builtin_amdgcn_wmma_f32_16x16x32_bf16(false, a_lo[ks], false, bh, (short)0, c, false, false);
      c = __builtin_amdgcn_wmma_f32_16x16x32_bf16(false, a_hi[ks], false, bl, (short)0, c, false, false);
    }
    const float bn = bias[n0 + mrow];
#pragma unroll
    for (int v = 0; v < 8; ++v) {
      float val = c[v] + bn;
      orow[(size_t)v * HIDDEN_N + n0] = fmaxf(val, 0.0f);
    }
  }
}

// ---------------- launcher ----------------

static inline size_t align256(size_t x) { return (x + 255) & ~(size_t)255; }

extern "C" void kernel_launch(void* const* d_in, const int* in_sizes, int n_in,
                              void* d_out, int out_size, void* d_ws, size_t ws_size,
                              hipStream_t stream) {
  const float* x    = (const float*)d_in[0];
  const int*   esrc = (const int*)d_in[1];
  const int*   edst = (const int*)d_in[2];
  const float* W    = (const float*)d_in[3];
  const float* b    = (const float*)d_in[4];
  float*       out  = (float*)d_out;

  const int n_nodes = in_sizes[0] / N_FEAT;   // 100000 (multiple of 16)
  const int n_edges = in_sizes[1];            // 800000

  // workspace layout (~116.4 MB total)
  char*  ws  = (char*)d_ws;
  size_t off = 0;
  float* ns  = (float*)(ws + off);                        // [N,96] neighbor sums
  float* deg = ns + (size_t)n_nodes * N_FEAT;             // [N] degrees (contiguous for zeroing)
  off += align256((size_t)n_nodes * (N_FEAT + 1) * sizeof(float));
  float* hfull = (float*)(ws + off);                      // [N,192] fp32 concat features
  off += align256((size_t)n_nodes * H_FEAT * sizeof(float));
  bf16_t* wh = (bf16_t*)(ws + off);                       // [1024,192] bf16 hi (transposed)
  off += align256((size_t)H_FEAT * HIDDEN_N * sizeof(bf16_t));
  bf16_t* wl = (bf16_t*)(ws + off);                       // [1024,192] bf16 lo (transposed)

  const int zn = n_nodes * (N_FEAT + 1);
  gnn_zero<<<(zn + 255) / 256, 256, 0, stream>>>(ns, zn);

  gnn_deg<<<(n_edges + 255) / 256, 256, 0, stream>>>(edst, deg, n_edges);

  const int st = n_edges * N_FEAT;   // 76.8M
  gnn_scatter<<<(st + 255) / 256, 256, 0, stream>>>(x, esrc, edst, ns, st);

  const int ft = n_nodes * N_FEAT;
  gnn_finalize<<<(ft + 255) / 256, 256, 0, stream>>>(x, ns, deg, hfull, ft);

  const int wt = H_FEAT * HIDDEN_N;
  gnn_wprep<<<(wt + 255) / 256, 256, 0, stream>>>(W, wh, wl, wt);

  const int mtiles = n_nodes / 16;   // 6250
  gnn_gemm<<<(mtiles + 7) / 8, 256, 0, stream>>>(hfull, wh, wl, b, out, mtiles);
}
